// FastMultiheadAttention_2860448219139
// MI455X (gfx1250) — matrix operand order repaired and checked by reading, not compiled, not run on hardware
//
#include <hip/hip_runtime.h>

typedef __attribute__((ext_vector_type(16))) __bf16 v16bf;
typedef __attribute__((ext_vector_type(8)))  __bf16 v8bf;
typedef __attribute__((ext_vector_type(8)))  float  v8f;

#define BM 128
#define BN 64
#define BK 32
#define BKP 40   // row stride 80B: 16B-aligned for async b128, conflict-free 20-dword stride

enum { LAY_MK = 0, LAY_KM = 1 };               // A global layout
enum { LAY_KN = 0, LAY_NK = 1 };               // B global layout (NK = torch weight [N][K])
enum { EPI_BF16_BIAS = 0, EPI_PHI = 1, EPI_ATOMIC = 2, EPI_DIV = 3, EPI_F32_BIAS = 4 };

// CDNA5 async copy: global -> LDS, 16B per lane, tracked by ASYNCcnt.
__device__ __forceinline__ void async_copy_b128(unsigned lds_byte_addr, const void* gptr) {
  asm volatile("global_load_async_to_lds_b128 %0, %1, off"
               :: "v"(lds_byte_addr), "v"(gptr) : "memory");
}
__device__ __forceinline__ void wait_asynccnt0() {
  asm volatile("s_wait_asynccnt 0x0" ::: "memory");
}
template <typename T>
__device__ __forceinline__ unsigned lds_addr32(T* p) {
  // LDS aperture: generic addr[31:0] is the LDS byte address (ISA 10.2)
  return (unsigned)(unsigned long long)(void*)p;
}

// Generic bf16 WMMA GEMM: C = A * B (+epilogue). 256 threads = 8 waves (wave32).
// Block tile 128x64, wave tile 32x32 (2x2 WMMA fragments), K step 32, double-buffered LDS.
template<int ALAY, int BLAY, int EPI, bool AF32, bool BF32>
__global__ __launch_bounds__(256) void k_gemm(
    const void* __restrict__ Ap, long long sAb, long long sAh, int ldA,
    const void* __restrict__ Bp, long long sBb, long long sBh, int ldB,
    void* __restrict__ Cp, long long sCb, long long sCh, int ldC,
    const float* __restrict__ bias,
    const float* __restrict__ aux, long long sAux,
    int K, int HB, int ksplit, int kchunk, float scale)
{
  constexpr bool A_ASYNC = (ALAY == LAY_MK) && !AF32;

  __shared__ __align__(16) __bf16 lA[2][BM][BKP];    // [buf][m][k]
  __shared__ __align__(16) __bf16 lBt[2][BN][BKP];   // [buf][n][k] (transposed)

  const int tid   = threadIdx.x;
  const int z     = blockIdx.z;
  const int batch = z / ksplit;
  const int ks    = z - batch * ksplit;
  const int bb    = batch / HB;
  const int bh    = batch - bb * HB;

  const long long offA = (long long)bb * sAb + (long long)bh * sAh;
  const long long offB = (long long)bb * sBb + (long long)bh * sBh;
  const long long offC = (long long)bb * sCb + (long long)bh * sCh;

  const int m0 = blockIdx.x * BM;
  const int n0 = blockIdx.y * BN;

  const int kbeg = ks * kchunk;
  int kend = kbeg + kchunk; if (kend > K) kend = K;

  const int lane = tid & 31;
  const int w    = tid >> 5;
  const int wm   = w & 3;            // 4 waves along M
  const int wn   = w >> 2;           // 2 waves along N

  v8f acc[2][2];
  const v8f zv = {0.f,0.f,0.f,0.f,0.f,0.f,0.f,0.f};
  #pragma unroll
  for (int i = 0; i < 2; ++i)
    #pragma unroll
    for (int j = 0; j < 2; ++j) acc[i][j] = zv;

  // ---- staging: one (BM x BK) A tile + (BK x BN) B tile into LDS buffer `buf` ----
  auto stage = [&](int kb, int buf) {
    // A tile
    if constexpr (ALAY == LAY_MK) {
      const int ml = tid >> 1, half = tid & 1;            // 128 rows x (2 x 16 elems)
      if constexpr (AF32) {
        const float* s = (const float*)Ap + offA + (long long)(m0 + ml) * ldA + kb + half * 16;
        float4 f0 = ((const float4*)s)[0];
        float4 f1 = ((const float4*)s)[1];
        float4 f2 = ((const float4*)s)[2];
        float4 f3 = ((const float4*)s)[3];
        __bf16* d = &lA[buf][ml][half * 16];
        d[0]=(__bf16)f0.x;  d[1]=(__bf16)f0.y;  d[2]=(__bf16)f0.z;  d[3]=(__bf16)f0.w;
        d[4]=(__bf16)f1.x;  d[5]=(__bf16)f1.y;  d[6]=(__bf16)f1.z;  d[7]=(__bf16)f1.w;
        d[8]=(__bf16)f2.x;  d[9]=(__bf16)f2.y;  d[10]=(__bf16)f2.z; d[11]=(__bf16)f2.w;
        d[12]=(__bf16)f3.x; d[13]=(__bf16)f3.y; d[14]=(__bf16)f3.z; d[15]=(__bf16)f3.w;
      } else {
        // CDNA5 async copy straight to LDS (2 x 16B per thread), no VGPR round trip
        const __bf16* s = (const __bf16*)Ap + offA + (long long)(m0 + ml) * ldA + kb + half * 16;
        const unsigned l0 = lds_addr32(&lA[buf][ml][half * 16]);
        async_copy_b128(l0, s);
        async_copy_b128(l0 + 16, s + 8);
      }
    } else {  // LAY_KM: global A is [K][M] row-major (e.g. Kp[n][m]); transpose into LDS
      const int kl = tid >> 3, seg = tid & 7;             // 32 k-rows x (8 x 16 elems)
      const __bf16* s = (const __bf16*)Ap + offA + (long long)(kb + kl) * ldA + m0 + seg * 16;
      v8bf u0 = ((const v8bf*)s)[0];
      v8bf u1 = ((const v8bf*)s)[1];
      #pragma unroll
      for (int j = 0; j < 8; ++j) {
        lA[buf][seg * 16 + j][kl] = u0[j];
        lA[buf][seg * 16 + 8 + j][kl] = u1[j];
      }
    }
    // B tile (always transposed into lBt[n][k])
    if constexpr (BLAY == LAY_KN) {
      const int kl = tid >> 3, seg = tid & 7;             // 32 k-rows x (8 x 8 elems)
      if constexpr (BF32) {
        const float* s = (const float*)Bp + offB + (long long)(kb + kl) * ldB + n0 + seg * 8;
        float4 f0 = ((const float4*)s)[0];
        float4 f1 = ((const float4*)s)[1];
        lBt[buf][seg*8+0][kl]=(__bf16)f0.x; lBt[buf][seg*8+1][kl]=(__bf16)f0.y;
        lBt[buf][seg*8+2][kl]=(__bf16)f0.z; lBt[buf][seg*8+3][kl]=(__bf16)f0.w;
        lBt[buf][seg*8+4][kl]=(__bf16)f1.x; lBt[buf][seg*8+5][kl]=(__bf16)f1.y;
        lBt[buf][seg*8+6][kl]=(__bf16)f1.z; lBt[buf][seg*8+7][kl]=(__bf16)f1.w;
      } else {
        const __bf16* s = (const __bf16*)Bp + offB + (long long)(kb + kl) * ldB + n0 + seg * 8;
        v8bf u = *(const v8bf*)s;
        #pragma unroll
        for (int j = 0; j < 8; ++j) lBt[buf][seg * 8 + j][kl] = u[j];
      }
    } else {  // LAY_NK: global B is [N][K] row-major (torch weights, proj)
      const int nl = tid >> 2, seg = tid & 3;             // 64 n-rows x (4 x 8 elems)
      if constexpr (BF32) {
        const float* s = (const float*)Bp + offB + (long long)(n0 + nl) * ldB + kb + seg * 8;
        float4 f0 = ((const float4*)s)[0];
        float4 f1 = ((const float4*)s)[1];
        __bf16* d = &lBt[buf][nl][seg * 8];
        d[0]=(__bf16)f0.x; d[1]=(__bf16)f0.y; d[2]=(__bf16)f0.z; d[3]=(__bf16)f0.w;
        d[4]=(__bf16)f1.x; d[5]=(__bf16)f1.y; d[6]=(__bf16)f1.z; d[7]=(__bf16)f1.w;
      } else {
        const __bf16* s = (const __bf16*)Bp + offB + (long long)(n0 + nl) * ldB + kb + seg * 8;
        v8bf u = *(const v8bf*)s;
        __bf16* d = &lBt[buf][nl][seg * 8];
        #pragma unroll
        for (int j = 0; j < 8; ++j) d[j] = u[j];
      }
    }
  };

  // ---- compute: gather fragments per ISA VGPR layouts + 4 WMMAs ----
  auto compute = [&](int buf) {
    v16bf afr[2], bfr[2];
    const int klo  = (lane & 16) ? 8  : 0;   // A: lanes16-31 hold K 8-15 / 24-31
    const int klo2 = klo;                    // B: same K order as A (lanes16-31 hold K 8-15 / 24-31)
    #pragma unroll
    for (int sm = 0; sm < 2; ++sm) {
      const int r = wm * 32 + sm * 16 + (lane & 15);
      #pragma unroll
      for (int v = 0; v < 8; ++v) {
        const int kk = (v < 4) ? (klo + 2 * v) : (16 + klo + 2 * (v - 4));
        afr[sm][2 * v]     = lA[buf][r][kk];
        afr[sm][2 * v + 1] = lA[buf][r][kk + 1];
      }
    }
    #pragma unroll
    for (int sn = 0; sn < 2; ++sn) {
      const int c = wn * 32 + sn * 16 + (lane & 15);
      #pragma unroll
      for (int v = 0; v < 8; ++v) {
        const int kk = (v < 4) ? (klo2 + 2 * v) : (16 + klo2 + 2 * (v - 4));
        bfr[sn][2 * v]     = lBt[buf][c][kk];
        bfr[sn][2 * v + 1] = lBt[buf][c][kk + 1];
      }
    }
    #pragma unroll
    for (int sm = 0; sm < 2; ++sm)
      #pragma unroll
      for (int sn = 0; sn < 2; ++sn)
        acc[sm][sn] = __builtin_amdgcn_wmma_f32_16x16x32_bf16(
            false, afr[sm], false, bfr[sn], (short)0, acc[sm][sn], false, false);
  };

  // ---- double-buffered main loop ----
  int cur = 0;
  stage(kbeg, 0);
  for (int kb = kbeg; kb < kend; kb += BK) {
    if constexpr (A_ASYNC) wait_asynccnt0();   // per-wave: asyncs into `cur` complete
    __syncthreads();                           // stage(cur) visible; prev reads of cur^1 done
    if (kb + BK < kend) stage(kb + BK, cur ^ 1);
    compute(cur);
    cur ^= 1;
  }

  // ---- epilogue (C layout: VGPR i -> M = i + 8*(lane>=16), N = lane&15) ----
  const int rb = (lane & 16) ? 8 : 0;
  const int cl = lane & 15;
  #pragma unroll
  for (int sm = 0; sm < 2; ++sm)
    #pragma unroll
    for (int sn = 0; sn < 2; ++sn)
      #pragma unroll
      for (int i = 0; i < 8; ++i) {
        const int row = m0 + wm * 32 + sm * 16 + rb + i;
        const int col = n0 + wn * 32 + sn * 16 + cl;
        float v = acc[sm][sn][i];
        const long long ci = offC + (long long)row * ldC + col;
        if constexpr (EPI == EPI_BF16_BIAS) {
          v = (v + bias[col]) * scale;
          ((__bf16*)Cp)[ci] = (__bf16)v;
        } else if constexpr (EPI == EPI_PHI) {
          v = 0.0625f * __expf(v - aux[(long long)batch * sAux + row]) + 1e-4f;
          ((__bf16*)Cp)[ci] = (__bf16)v;
        } else if constexpr (EPI == EPI_ATOMIC) {
          atomicAdd((float*)Cp + ci, v);
        } else if constexpr (EPI == EPI_DIV) {
          v = v / aux[(long long)batch * sAux + row];
          ((__bf16*)Cp)[ci] = (__bf16)v;
        } else {
          ((float*)Cp)[ci] = v + bias[col];
        }
      }
}

__global__ __launch_bounds__(256) void k_zero(float* p, int n) {
  int i = blockIdx.x * 256 + threadIdx.x;
  if (i < n) p[i] = 0.f;
}

// diag[b,h,n] = 0.5 * sum_d x^2, x = scaled projected Q/K (bf16, [b][n][512] head slices)
__global__ __launch_bounds__(256) void k_diag(const __bf16* __restrict__ Xs,
                                              float* __restrict__ diag) {
  int gid = blockIdx.x * 256 + threadIdx.x;      // 16*8192 rows
  int bt  = gid >> 13;
  int n   = gid & 8191;
  int bb  = bt >> 3, bh = bt & 7;
  const __bf16* s = Xs + ((long long)(bb * 8192 + n)) * 512 + bh * 64;
  float acc = 0.f;
  #pragma unroll
  for (int j = 0; j < 8; ++j) {
    v8bf u = ((const v8bf*)s)[j];
    #pragma unroll
    for (int e = 0; e < 8; ++e) { float f = (float)u[e]; acc += f * f; }
  }
  diag[gid] = 0.5f * acc;
}

// Ksum[batch][m] += column sums of Kp over 1024-row chunks
__global__ __launch_bounds__(256) void k_ksum(const __bf16* __restrict__ Kp,
                                              float* __restrict__ Ksum) {
  int batch = blockIdx.x, chunk = blockIdx.y, m = threadIdx.x;
  const __bf16* s = Kp + (long long)batch * 8192 * 256 + (long long)chunk * 1024 * 256 + m;
  float acc = 0.f;
  for (int r = 0; r < 1024; ++r) acc += (float)s[(long long)r * 256];
  atomicAdd(&Ksum[batch * 256 + m], acc);
}

__global__ __launch_bounds__(256) void k_cvt(const float* __restrict__ s,
                                             __bf16* __restrict__ d, int n) {
  int i = blockIdx.x * 256 + threadIdx.x;
  if (i < n) d[i] = (__bf16)s[i];
}

// denom[batch][r] = dot(Qp[r,:], Ksum[batch,:])
__global__ __launch_bounds__(256) void k_denom(const __bf16* __restrict__ Qp,
                                               const float* __restrict__ Ksum,
                                               float* __restrict__ denom) {
  __shared__ float sk[256];
  int batch = blockIdx.x, t = threadIdx.x;
  sk[t] = Ksum[batch * 256 + t];
  __syncthreads();
  int r = blockIdx.y * 256 + t;
  const v8bf* q = (const v8bf*)(Qp + (long long)batch * 8192 * 256 + (long long)r * 256);
  float acc = 0.f;
  #pragma unroll 4
  for (int j = 0; j < 32; ++j) {
    v8bf u = q[j];
    #pragma unroll
    for (int e = 0; e < 8; ++e) acc += (float)u[e] * sk[j * 8 + e];
  }
  denom[batch * 8192 + r] = acc;
}

extern "C" void kernel_launch(void* const* d_in, const int* in_sizes, int n_in,
                              void* d_out, int out_size, void* d_ws, size_t ws_size,
                              hipStream_t stream) {
  (void)in_sizes; (void)n_in; (void)out_size; (void)ws_size;
  const float* query = (const float*)d_in[0];
  const float* key_  = (const float*)d_in[1];
  const float* value = (const float*)d_in[2];
  const float* WQw = (const float*)d_in[3];
  const float* WQb = (const float*)d_in[4];
  const float* WKw = (const float*)d_in[5];
  const float* WKb = (const float*)d_in[6];
  const float* WVw = (const float*)d_in[7];
  const float* WVb = (const float*)d_in[8];
  const float* WOw = (const float*)d_in[9];
  const float* WOb = (const float*)d_in[10];
  const float* proj = (const float*)d_in[11];
  float* out = (float*)d_out;

  char* ws = (char*)d_ws;
  size_t off = 0;
  auto take = [&](size_t bytes) -> char* {
    char* p = ws + off;
    off = (off + bytes + 255) & ~(size_t)255;
    return p;
  };

  __bf16* Qs    = (__bf16*)take(16384ull * 512 * 2);      // scaled projected Q (bf16)
  __bf16* Ks    = (__bf16*)take(16384ull * 512 * 2);
  __bf16* Vs    = (__bf16*)take(16384ull * 512 * 2);
  __bf16* Qp    = (__bf16*)take(16ull * 8192 * 256 * 2);  // phi(Q)
  __bf16* Kp    = (__bf16*)take(16ull * 8192 * 256 * 2);  // phi(K)
  float*  diagQ = (float*)take(131072ull * 4);
  float*  diagK = (float*)take(131072ull * 4);
  float*  ctxf  = (float*)take(16ull * 256 * 64 * 4);     // split-K accumulator
  float*  Ksum  = (float*)take(16ull * 256 * 4);          // contiguous after ctxf
  __bf16* ctxb  = (__bf16*)take(16ull * 256 * 64 * 2);
  float*  denom = (float*)take(131072ull * 4);
  __bf16* attnb = (__bf16*)take(16384ull * 512 * 2);

  const float S = 0.35355339059327373f;  // 64^(-0.25)

  // zero ctxf + Ksum (contiguous: 262144 + 4096 floats)
  k_zero<<<dim3(1040), 256, 0, stream>>>(ctxf, 266240);

  // QKV projections: C = (A @ W^T + b) * scale  -> bf16
  k_gemm<LAY_MK, LAY_NK, EPI_BF16_BIAS, true, true><<<dim3(128, 8, 1), 256, 0, stream>>>(
      query, 0, 0, 512, WQw, 0, 0, 512, Qs, 0, 0, 512, WQb, nullptr, 0, 512, 1, 1, 512, S);
  k_gemm<LAY_MK, LAY_NK, EPI_BF16_BIAS, true, true><<<dim3(128, 8, 1), 256, 0, stream>>>(
      key_, 0, 0, 512, WKw, 0, 0, 512, Ks, 0, 0, 512, WKb, nullptr, 0, 512, 1, 1, 512, S);
  k_gemm<LAY_MK, LAY_NK, EPI_BF16_BIAS, true, true><<<dim3(128, 8, 1), 256, 0, stream>>>(
      value, 0, 0, 512, WVw, 0, 0, 512, Vs, 0, 0, 512, WVb, nullptr, 0, 512, 1, 1, 512, 1.f);

  k_diag<<<dim3(512), 256, 0, stream>>>(Qs, diagQ);
  k_diag<<<dim3(512), 256, 0, stream>>>(Ks, diagK);

  // dash + phi epilogue: Qp/Kp = exp(x @ proj^T - diag)/16 + eps   (16 head batches)
  k_gemm<LAY_MK, LAY_NK, EPI_PHI, false, true><<<dim3(64, 4, 16), 256, 0, stream>>>(
      Qs, 4194304, 64, 512, proj, 0, 0, 64, Qp, 16777216, 2097152, 256,
      nullptr, diagQ, 8192, 64, 8, 1, 64, 1.f);
  k_gemm<LAY_MK, LAY_NK, EPI_PHI, false, true><<<dim3(64, 4, 16), 256, 0, stream>>>(
      Ks, 4194304, 64, 512, proj, 0, 0, 64, Kp, 16777216, 2097152, 256,
      nullptr, diagK, 8192, 64, 8, 1, 64, 1.f);

  k_ksum<<<dim3(16, 8), 256, 0, stream>>>(Kp, Ksum);

  // ctx = Kp^T @ V  (M=256,N=64,K=8192), split-K=8, atomic f32 accumulate
  k_gemm<LAY_KM, LAY_KN, EPI_ATOMIC, false, false><<<dim3(2, 1, 128), 256, 0, stream>>>(
      Kp, 16777216, 2097152, 256, Vs, 4194304, 64, 512, ctxf, 131072, 16384, 64,
      nullptr, nullptr, 0, 8192, 8, 8, 1024, 1.f);

  k_cvt<<<dim3(1024), 256, 0, stream>>>(ctxf, ctxb, 262144);

  k_denom<<<dim3(16, 32), 256, 0, stream>>>(Qp, Ksum, denom);

  // attn = (Qp @ ctx) / denom -> bf16
  k_gemm<LAY_MK, LAY_KN, EPI_DIV, false, false><<<dim3(64, 1, 16), 256, 0, stream>>>(
      Qp, 16777216, 2097152, 256, ctxb, 131072, 16384, 64, attnb, 4194304, 524288, 64,
      nullptr, denom, 8192, 256, 8, 1, 256, 1.f);

  // out = attn_flat @ WO^T + b -> f32
  k_gemm<LAY_MK, LAY_NK, EPI_F32_BIAS, false, true><<<dim3(128, 8, 1), 256, 0, stream>>>(
      attnb, 0, 0, 512, WOw, 0, 0, 512, out, 0, 0, 512, WOb, nullptr, 0, 512, 1, 1, 512, 1.f);
}
